// Local2TransformerBlock_2740189134928
// MI455X (gfx1250) — matrix operand order repaired and checked by reading, not compiled, not run on hardware
//
#include <hip/hip_runtime.h>
#include <hip/hip_bf16.h>

typedef __bf16 bf16_t;
typedef __attribute__((ext_vector_type(16))) __bf16 v16bf;
typedef __attribute__((ext_vector_type(8)))  float  v8f;

#define BB   4
#define TT   4096
#define DD   1024
#define HH   8
#define DHH  128
#define MLPD 4096
#define MM   (BB * TT)      // 16384 tokens

// ---------------------------------------------------------------------------
// WMMA fragment loaders (contiguous-K order used here for both A and B, wave32).
// A (16x32, MxK) row-major: lane m=l&15, half=l>>4 holds K = 16h..16h+15
// B (32x16, KxN) stored as [N,K] row-major: lane n=l&15, half h holds K = 16h..16h+15
// ---------------------------------------------------------------------------
union FragU { uint4 u[2]; v16bf v; };

__device__ __forceinline__ v16bf load_frag_a(const bf16_t* base, int lda) {
  const int lane = threadIdx.x & 31;
  const int m = lane & 15, hf = lane >> 4;
  const bf16_t* p = base + (size_t)m * lda + 16 * hf;
  FragU f;
  f.u[0] = *(const uint4*)(p);        // K = 16h .. 16h+7
  f.u[1] = *(const uint4*)(p + 8);    // K = 16h+8 .. 16h+15
  return f.v;
}

__device__ __forceinline__ v16bf load_frag_bT(const bf16_t* base, int ldb) {
  const int lane = threadIdx.x & 31;
  const int n = lane & 15, hf = lane >> 4;
  const bf16_t* p = base + (size_t)n * ldb + 16 * hf;   // 16 contiguous K
  FragU f;
  f.u[0] = *(const uint4*)(p);
  f.u[1] = *(const uint4*)(p + 8);
  return f.v;
}

__device__ __forceinline__ v8f wmma_bf16(v16bf a, v16bf b, v8f c) {
  return __builtin_amdgcn_wmma_f32_16x16x32_bf16(false, a, false, b, (short)0, c,
                                                 false, false);
}

// ---------------------------------------------------------------------------
// Weight convert + transpose: fp32 [K,N] row-major -> bf16 [N,K] row-major
// ---------------------------------------------------------------------------
__global__ __launch_bounds__(256) void convT_kernel(const float* __restrict__ in,
                                                    bf16_t* __restrict__ out,
                                                    int K, int N) {
  int idx = blockIdx.x * 256 + threadIdx.x;
  if (idx >= K * N) return;
  int k = idx / N, n = idx - k * N;
  out[(size_t)n * K + k] = (bf16_t)in[idx];
}

// ---------------------------------------------------------------------------
// LayerNorm (+ optional sinusoidal PE), fp32 in -> bf16 out. One block/token.
// ---------------------------------------------------------------------------
__device__ __forceinline__ float sin_pe(int pos, int d) {
  const float log_inc = 9.210340371976184f / 511.0f;  // ln(10000)/(D/2-1)
  int j = (d < 512) ? d : (d - 512);
  float ang = (float)pos * __expf(-log_inc * (float)j);
  return (d < 512) ? __sinf(ang) : __cosf(ang);
}

__global__ __launch_bounds__(256) void ln_kernel(const float* __restrict__ x,
                                                 const float* __restrict__ sc,
                                                 const float* __restrict__ bi,
                                                 bf16_t* __restrict__ out,
                                                 int addPE) {
  const int tok = blockIdx.x;
  const int tid = threadIdx.x;
  const float* xp = x + (size_t)tok * DD;
  float4 v = *(const float4*)(xp + tid * 4);

  float s = v.x + v.y + v.z + v.w;
#pragma unroll
  for (int o = 1; o < 32; o <<= 1) s += __shfl_xor(s, o, 32);
  __shared__ float red[8];
  __shared__ float stats[2];
  const int wid = tid >> 5, lane = tid & 31;
  if (lane == 0) red[wid] = s;
  __syncthreads();
  if (tid == 0) {
    float t = 0.f;
#pragma unroll
    for (int i = 0; i < 8; i++) t += red[i];
    stats[0] = t * (1.0f / DD);
  }
  __syncthreads();
  const float mu = stats[0];
  float d0 = v.x - mu, d1 = v.y - mu, d2 = v.z - mu, d3 = v.w - mu;
  float ss = d0 * d0 + d1 * d1 + d2 * d2 + d3 * d3;
#pragma unroll
  for (int o = 1; o < 32; o <<= 1) ss += __shfl_xor(ss, o, 32);
  __syncthreads();
  if (lane == 0) red[wid] = ss;
  __syncthreads();
  if (tid == 0) {
    float t = 0.f;
#pragma unroll
    for (int i = 0; i < 8; i++) t += red[i];
    stats[1] = rsqrtf(t * (1.0f / DD) + 1e-6f);
  }
  __syncthreads();
  const float rstd = stats[1];
  const int pos = tok & (TT - 1);
  bf16_t* op = out + (size_t)tok * DD;
  float e[4] = {d0, d1, d2, d3};
#pragma unroll
  for (int i = 0; i < 4; i++) {
    int d = tid * 4 + i;
    float val = e[i] * rstd * sc[d] + bi[d];
    if (addPE) val += sin_pe(pos, d);
    op[d] = (bf16_t)val;
  }
}

// ---------------------------------------------------------------------------
// Generic bf16 WMMA GEMM: C[M,N] = A[M,K] * B^T[N,K], with epilogue modes.
// 256 threads = 8 waves (2x4), block tile 128x256, 64x64 per wave.
// ---------------------------------------------------------------------------
#define EPI_SCALE_BF16  0   // outB = v*scale          (Q / K projections)
#define EPI_VT          1   // outB transposed [B,H*DH,T] (V projection)
#define EPI_RESID_F32   2   // outF = v + resid        (attn out-proj)
#define EPI_GELU_BF16   3   // outB = gelu(v + bias)   (MLP up)
#define EPI_BIAS_RES_F32 4  // outF = v + bias + resid (MLP down -> d_out)

template <int MODE>
__global__ __launch_bounds__(256) void gemm_bf16_kernel(
    const bf16_t* __restrict__ A, const bf16_t* __restrict__ B,
    float* __restrict__ outF, bf16_t* __restrict__ outB,
    const float* __restrict__ bias, const float* __restrict__ resid,
    int M, int N, int K, float scale) {
  const int lane = threadIdx.x & 31;
  const int wid = threadIdx.x >> 5;
  const int l16 = lane & 15, hf = lane >> 4;
  const int row0 = blockIdx.y * 128 + (wid >> 2) * 64;
  const int col0 = blockIdx.x * 256 + (wid & 3) * 64;

  const v8f vz = {0.f, 0.f, 0.f, 0.f, 0.f, 0.f, 0.f, 0.f};
  v8f acc[4][4];
#pragma unroll
  for (int i = 0; i < 4; i++)
#pragma unroll
    for (int j = 0; j < 4; j++) acc[i][j] = vz;

  for (int k0 = 0; k0 < K; k0 += 32) {
    v16bf af[4], bfv[4];
#pragma unroll
    for (int i = 0; i < 4; i++)
      af[i] = load_frag_a(A + (size_t)(row0 + 16 * i) * K + k0, K);
#pragma unroll
    for (int j = 0; j < 4; j++)
      bfv[j] = load_frag_bT(B + (size_t)(col0 + 16 * j) * K + k0, K);
    if (k0 + 32 < K) {  // speculative prefetch of next A K-slab
      __builtin_prefetch(A + (size_t)row0 * K + k0 + 32, 0, 1);
      __builtin_prefetch(B + (size_t)col0 * K + k0 + 32, 0, 1);
    }
#pragma unroll
    for (int i = 0; i < 4; i++)
#pragma unroll
      for (int j = 0; j < 4; j++) acc[i][j] = wmma_bf16(af[i], bfv[j], acc[i][j]);
  }

#pragma unroll
  for (int i = 0; i < 4; i++)
#pragma unroll
    for (int j = 0; j < 4; j++)
#pragma unroll
      for (int r = 0; r < 8; r++) {
        const int row = row0 + 16 * i + r + 8 * hf;
        const int col = col0 + 16 * j + l16;
        const float v = acc[i][j][r];
        const size_t idx = (size_t)row * N + col;
        if (MODE == EPI_SCALE_BF16) {
          outB[idx] = (bf16_t)(v * scale);
        } else if (MODE == EPI_VT) {
          const int b = row >> 12;              // row / T (T = 4096)
          const int t = row & (TT - 1);
          outB[((size_t)b * N + col) * TT + t] = (bf16_t)v;
        } else if (MODE == EPI_RESID_F32) {
          outF[idx] = v + resid[idx];
        } else if (MODE == EPI_GELU_BF16) {
          const float u = v + bias[col];
          const float th = tanhf(0.7978845608028654f * (u + 0.044715f * u * u * u));
          outB[idx] = (bf16_t)(0.5f * u * (1.f + th));
        } else {  // EPI_BIAS_RES_F32
          outF[idx] = v + bias[col] + resid[idx];
        }
      }
}

// ---------------------------------------------------------------------------
// Block-local attention: one wave per (batch, 64-token block, head).
// scores = Q Kt (64x64, K=128) -> mask -> softmax (regs) -> LDS -> P*V (N=128)
// ---------------------------------------------------------------------------
__global__ __launch_bounds__(32) void attn_kernel(
    const bf16_t* __restrict__ q, const bf16_t* __restrict__ kmat,
    const bf16_t* __restrict__ vT, const unsigned char* __restrict__ mask,
    bf16_t* __restrict__ ctx) {
  const int NB = TT / 64;
  const int h = blockIdx.x % HH;
  const int nbi = (blockIdx.x / HH) % NB;
  const int b = blockIdx.x / (HH * NB);
  const int HD = HH * DHH;           // 1024
  const int tok0 = b * TT + nbi * 64;
  const bf16_t* qp = q + (size_t)tok0 * HD + h * DHH;
  const bf16_t* kp = kmat + (size_t)tok0 * HD + h * DHH;

  const int lane = threadIdx.x & 31;
  const int l16 = lane & 15, hf = lane >> 4;

  const v8f vz = {0.f, 0.f, 0.f, 0.f, 0.f, 0.f, 0.f, 0.f};
  v8f sc[4][4];
#pragma unroll
  for (int i = 0; i < 4; i++)
#pragma unroll
    for (int j = 0; j < 4; j++) sc[i][j] = vz;

  for (int k0 = 0; k0 < DHH; k0 += 32) {
    v16bf af[4], bfv[4];
#pragma unroll
    for (int i = 0; i < 4; i++)
      af[i] = load_frag_a(qp + (size_t)(16 * i) * HD + k0, HD);
#pragma unroll
    for (int j = 0; j < 4; j++)          // K rows are the WMMA [N,K] operand
      bfv[j] = load_frag_bT(kp + (size_t)(16 * j) * HD + k0, HD);
#pragma unroll
    for (int i = 0; i < 4; i++)
#pragma unroll
      for (int j = 0; j < 4; j++) sc[i][j] = wmma_bf16(af[i], bfv[j], sc[i][j]);
  }

  // key padding mask
  const unsigned char* mp = mask + (size_t)b * TT + nbi * 64;
  float mb[4];
#pragma unroll
  for (int j = 0; j < 4; j++) mb[j] = mp[16 * j + l16] ? 0.f : -1e9f;
#pragma unroll
  for (int i = 0; i < 4; i++)
#pragma unroll
    for (int j = 0; j < 4; j++)
#pragma unroll
      for (int r = 0; r < 8; r++) sc[i][j][r] += mb[j];

  // softmax over each row (row lives in one VGPR across 16 lanes of one half)
  __shared__ bf16_t sh[64 * 80];       // padded stride: 160B rows, b128-aligned
#pragma unroll
  for (int i = 0; i < 4; i++)
#pragma unroll
    for (int r = 0; r < 8; r++) {
      float mx = -3.4e38f;
#pragma unroll
      for (int j = 0; j < 4; j++) mx = fmaxf(mx, sc[i][j][r]);
#pragma unroll
      for (int o = 1; o < 16; o <<= 1) mx = fmaxf(mx, __shfl_xor(mx, o, 32));
      float e[4], sum = 0.f;
#pragma unroll
      for (int j = 0; j < 4; j++) {
        e[j] = __expf(sc[i][j][r] - mx);
        sum += e[j];
      }
#pragma unroll
      for (int o = 1; o < 16; o <<= 1) sum += __shfl_xor(sum, o, 32);
      const float inv = 1.0f / sum;
      const int row = 16 * i + r + 8 * hf;
#pragma unroll
      for (int j = 0; j < 4; j++)
        sh[row * 80 + 16 * j + l16] = (bf16_t)(e[j] * inv);
    }
  __syncthreads();

  // ctx = P (64x64) * V (64x128); V stored transposed [B,H,DH,T] -> [N,K] ready
  const bf16_t* vp = vT + ((size_t)(b * HH + h) * DHH) * TT + nbi * 64;
  bf16_t* cp = ctx + (size_t)tok0 * HD + h * DHH;
#pragma unroll
  for (int nc = 0; nc < 2; nc++) {
    v8f c2[4][4];
#pragma unroll
    for (int i = 0; i < 4; i++)
#pragma unroll
      for (int j = 0; j < 4; j++) c2[i][j] = vz;
#pragma unroll
    for (int k0 = 0; k0 < 64; k0 += 32) {
      v16bf af[4], bfv[4];
#pragma unroll
      for (int i = 0; i < 4; i++)
        af[i] = load_frag_a(sh + (16 * i) * 80 + k0, 80);
#pragma unroll
      for (int j = 0; j < 4; j++)
        bfv[j] = load_frag_bT(vp + (size_t)(nc * 64 + 16 * j) * TT + k0, TT);
#pragma unroll
      for (int i = 0; i < 4; i++)
#pragma unroll
        for (int j = 0; j < 4; j++) c2[i][j] = wmma_bf16(af[i], bfv[j], c2[i][j]);
    }
#pragma unroll
    for (int i = 0; i < 4; i++)
#pragma unroll
      for (int j = 0; j < 4; j++)
#pragma unroll
        for (int r = 0; r < 8; r++) {
          const int row = 16 * i + r + 8 * hf;
          const int col = nc * 64 + 16 * j + l16;
          cp[(size_t)row * HD + col] = (bf16_t)c2[i][j][r];
        }
  }
}

// ---------------------------------------------------------------------------
extern "C" void kernel_launch(void* const* d_in, const int* in_sizes, int n_in,
                              void* d_out, int out_size, void* d_ws, size_t ws_size,
                              hipStream_t stream) {
  const float* inputs = (const float*)d_in[0];
  const unsigned char* pmask = (const unsigned char*)d_in[1];
  const float* ln1s = (const float*)d_in[2];
  const float* ln1b = (const float*)d_in[3];
  const float* Wq = (const float*)d_in[4];
  const float* Wk = (const float*)d_in[5];
  const float* Wv = (const float*)d_in[6];
  const float* Wo = (const float*)d_in[7];
  const float* ln2s = (const float*)d_in[8];
  const float* ln2b = (const float*)d_in[9];
  const float* W1 = (const float*)d_in[10];
  const float* b1 = (const float*)d_in[11];
  const float* W2 = (const float*)d_in[12];
  const float* b2 = (const float*)d_in[13];
  float* out = (float*)d_out;

  char* w = (char*)d_ws;
  auto carve = [&](size_t bytes) {
    void* p = (void*)w;
    w += (bytes + 255) & ~(size_t)255;
    return p;
  };
  bf16_t* wqT = (bf16_t*)carve((size_t)DD * DD * 2);
  bf16_t* wkT = (bf16_t*)carve((size_t)DD * DD * 2);
  bf16_t* wvT = (bf16_t*)carve((size_t)DD * DD * 2);
  bf16_t* woT = (bf16_t*)carve((size_t)DD * DD * 2);
  bf16_t* w1T = (bf16_t*)carve((size_t)DD * MLPD * 2);
  bf16_t* w2T = (bf16_t*)carve((size_t)MLPD * DD * 2);
  bf16_t* xln = (bf16_t*)carve((size_t)MM * DD * 2);
  bf16_t* qb  = (bf16_t*)carve((size_t)MM * DD * 2);
  bf16_t* kb  = (bf16_t*)carve((size_t)MM * DD * 2);
  bf16_t* vTb = (bf16_t*)carve((size_t)MM * DD * 2);
  bf16_t* ctx = (bf16_t*)carve((size_t)MM * DD * 2);
  float*  xres = (float*)carve((size_t)MM * DD * 4);
  bf16_t* yb  = (bf16_t*)carve((size_t)MM * DD * 2);
  bf16_t* hb  = (bf16_t*)carve((size_t)MM * MLPD * 2);

  // 1) weight downconvert + transpose to [N,K]
  convT_kernel<<<(DD * DD + 255) / 256, 256, 0, stream>>>(Wq, wqT, DD, DD);
  convT_kernel<<<(DD * DD + 255) / 256, 256, 0, stream>>>(Wk, wkT, DD, DD);
  convT_kernel<<<(DD * DD + 255) / 256, 256, 0, stream>>>(Wv, wvT, DD, DD);
  convT_kernel<<<(DD * DD + 255) / 256, 256, 0, stream>>>(Wo, woT, DD, DD);
  convT_kernel<<<(DD * MLPD + 255) / 256, 256, 0, stream>>>(W1, w1T, DD, MLPD);
  convT_kernel<<<(MLPD * DD + 255) / 256, 256, 0, stream>>>(W2, w2T, MLPD, DD);

  // 2) LN1 + positional encoding -> bf16
  ln_kernel<<<MM, 256, 0, stream>>>(inputs, ln1s, ln1b, xln, 1);

  // 3) Q/K/V projections (Q pre-scaled by 1/sqrt(DH); V stored transposed)
  dim3 gD(DD / 256, MM / 128);
  gemm_bf16_kernel<EPI_SCALE_BF16><<<gD, 256, 0, stream>>>(
      xln, wqT, nullptr, qb, nullptr, nullptr, MM, DD, DD, 0.08838834764831845f);
  gemm_bf16_kernel<EPI_SCALE_BF16><<<gD, 256, 0, stream>>>(
      xln, wkT, nullptr, kb, nullptr, nullptr, MM, DD, DD, 1.0f);
  gemm_bf16_kernel<EPI_VT><<<gD, 256, 0, stream>>>(
      xln, wvT, nullptr, vTb, nullptr, nullptr, MM, DD, DD, 1.0f);

  // 4) block-local attention
  attn_kernel<<<BB * (TT / 64) * HH, 32, 0, stream>>>(qb, kb, vTb, pmask, ctx);

  // 5) output projection + residual (fp32)
  gemm_bf16_kernel<EPI_RESID_F32><<<gD, 256, 0, stream>>>(
      ctx, woT, xres, nullptr, nullptr, inputs, MM, DD, DD, 1.0f);

  // 6) LN2 -> bf16
  ln_kernel<<<MM, 256, 0, stream>>>(xres, ln2s, ln2b, yb, 0);

  // 7) MLP up + gelu
  dim3 gU(MLPD / 256, MM / 128);
  gemm_bf16_kernel<EPI_GELU_BF16><<<gU, 256, 0, stream>>>(
      yb, w1T, nullptr, hb, b1, nullptr, MM, MLPD, DD, 1.0f);

  // 8) MLP down + bias + residual -> d_out
  gemm_bf16_kernel<EPI_BIAS_RES_F32><<<gD, 256, 0, stream>>>(
      hb, w2T, out, nullptr, b2, xres, MM, DD, MLPD, 1.0f);

  (void)in_sizes; (void)n_in; (void)out_size; (void)ws_size;
}
